// SOM_71150428225848
// MI455X (gfx1250) — matrix operand order repaired and checked by reading, not compiled, not run on hardware
//
#include <hip/hip_runtime.h>
#include <math.h>

#define DIM0 48
#define DIM1 48
#define GRID_G 2304            // 48*48
#define BATCH 32768
#define FEAT 128
#define ROWS_PER_BLK 16
#define WAVES 8
#define COL_TILES (GRID_G / 16)   // 144
#define DSTRIDE 2308              // 2304 + 4 pad: 8*DSTRIDE % 64 != 0 -> fewer bank conflicts

typedef __attribute__((ext_vector_type(16))) __bf16 v16bf;
typedef __attribute__((ext_vector_type(8)))  float  v8f;
typedef __attribute__((ext_vector_type(4)))  unsigned int v4u;

union V16 { v16bf v; v4u q[2]; };

__device__ __forceinline__ unsigned short f2bf(float f) {
  unsigned u = __float_as_uint(f);
  u += 0x7fffu + ((u >> 16) & 1u);        // round-to-nearest-even
  return (unsigned short)(u >> 16);
}

__device__ __forceinline__ unsigned long long shflxor_u64(unsigned long long v, int m) {
  unsigned lo = (unsigned)v, hi = (unsigned)(v >> 32);
  lo = (unsigned)__shfl_xor((int)lo, m, 32);
  hi = (unsigned)__shfl_xor((int)hi, m, 32);
  return ((unsigned long long)hi << 32) | lo;
}

// ---- prep: x -> bf16 (row-major), per-row sum of squares ----
__global__ void som_prep_x(const float* __restrict__ x, unsigned short* __restrict__ xb,
                           float* __restrict__ x2) {
  const int b = blockIdx.x, f = threadIdx.x;
  float v = x[b * FEAT + f];
  xb[b * FEAT + f] = f2bf(v);
  __shared__ float s[FEAT];
  s[f] = v * v;
  __syncthreads();
  for (int st = FEAT / 2; st > 0; st >>= 1) {
    if (f < st) s[f] += s[f + st];
    __syncthreads();
  }
  if (f == 0) x2[b] = s[0];
}

// ---- prep: w (F x G) -> wT bf16 (G x F), per-col sum of squares ----
__global__ void som_prep_w(const float* __restrict__ w, unsigned short* __restrict__ wt,
                           float* __restrict__ w2) {
  const int g = blockIdx.x, f = threadIdx.x;
  float v = w[f * GRID_G + g];
  wt[g * FEAT + f] = f2bf(v);
  __shared__ float s[FEAT];
  s[f] = v * v;
  __syncthreads();
  for (int st = FEAT / 2; st > 0; st >>= 1) {
    if (f < st) s[f] += s[f + st];
    __syncthreads();
  }
  if (f == 0) w2[g] = s[0];
}

__global__ void som_zero(float* p) { p[0] = 0.0f; }

// Load the 16x128 A block for this wave (rows rowBase..rowBase+15), bf16 WMMA layout.
__device__ __forceinline__ void load_A(const unsigned short* __restrict__ xb,
                                       int rowBase, int lmod, int half, v16bf a[4]) {
  const unsigned short* rp = xb + (size_t)(rowBase + lmod) * FEAT;
#pragma unroll
  for (int kk = 0; kk < 4; ++kk) {
    const int k0 = kk * 32;
    V16 t;
    t.q[0] = *(const v4u*)(rp + k0 + 8 * half);       // K = k0+8h .. +7
    t.q[1] = *(const v4u*)(rp + k0 + 16 + 8 * half);  // K = k0+16+8h .. +7
    a[kk] = t.v;
  }
}

// ---- fused pass: WMMA distances -> LDS tile, argmin, Gaussian-weighted loss ----
__global__ void __launch_bounds__(256) som_fused(
    const unsigned short* __restrict__ xb, const unsigned short* __restrict__ wt,
    const float* __restrict__ x2, const float* __restrict__ w2,
    int* __restrict__ minpos, float* __restrict__ lsum) {
  extern __shared__ float dl[];           // [ROWS_PER_BLK][DSTRIDE] distance tile
  const int tid  = threadIdx.x;
  const int wave = tid >> 5;
  const int lane = tid & 31;
  const int half = lane >> 4;
  const int lmod = lane & 15;
  const int rowBase = blockIdx.x * ROWS_PER_BLK;

  __shared__ float x2s[ROWS_PER_BLK];
  __shared__ unsigned long long red[ROWS_PER_BLK][WAVES];
  __shared__ float bi[ROWS_PER_BLK], bj[ROWS_PER_BLK];
  __shared__ float rs[256];

  if (tid < ROWS_PER_BLK) x2s[tid] = x2[rowBase + tid];
  __syncthreads();

  v16bf a[4];
  load_A(xb, rowBase, lmod, half, a);

  unsigned long long best[8];
#pragma unroll
  for (int r = 0; r < 8; ++r) best[r] = ~0ULL;

  for (int t = wave; t < COL_TILES; t += WAVES) {
    const int col = t * 16 + lmod;
    const unsigned short* cp = wt + (size_t)col * FEAT;

    // Issue the whole B tile (8 x b128) + w2 up front -> one clause,
    // progressive s_wait_loadcnt lets WMMA k overlap loads for k+1..3.
    V16 bb[4];
#pragma unroll
    for (int kk = 0; kk < 4; ++kk) {
      const int k0 = kk * 32 + 8 * half;
      bb[kk].q[0] = *(const v4u*)(cp + k0);
      bb[kk].q[1] = *(const v4u*)(cp + k0 + 16);
    }
    const float w2c = w2[col];

    v8f acc = {};
#pragma unroll
    for (int kk = 0; kk < 4; ++kk)
      acc = __builtin_amdgcn_wmma_f32_16x16x32_bf16(false, a[kk], false, bb[kk].v,
                                                    (short)0, acc, false, false);

#pragma unroll
    for (int r = 0; r < 8; ++r) {
      const int m16 = r + 8 * half;
      float d2 = x2s[m16] + w2c - 2.0f * acc[r];
      float dist = __builtin_amdgcn_sqrtf(fmaxf(d2, 0.0f));   // raw v_sqrt_f32
      dl[m16 * DSTRIDE + col] = dist;
      unsigned long long p =
          ((unsigned long long)__float_as_uint(dist) << 32) | (unsigned)col;
      if (p < best[r]) best[r] = p;                            // v_min_u64
    }
  }

  // argmin reduce across the 16 lanes of each half (cols of this wave's tiles)
#pragma unroll
  for (int r = 0; r < 8; ++r) {
    unsigned long long p = best[r];
#pragma unroll
    for (int m = 1; m <= 8; m <<= 1) {
      unsigned long long o = shflxor_u64(p, m);
      p = (o < p) ? o : p;
    }
    best[r] = p;
  }
  if (lmod == 0) {
#pragma unroll
    for (int r = 0; r < 8; ++r) red[r + 8 * half][wave] = best[r];
  }
  __syncthreads();   // also makes all dl[] writes visible
  if (tid < ROWS_PER_BLK) {
    unsigned long long p = red[tid][0];
#pragma unroll
    for (int wv = 1; wv < WAVES; ++wv) {
      unsigned long long o = red[tid][wv];
      p = (o < p) ? o : p;
    }
    const int mp = (int)(p & 0xffffffffu);
    minpos[rowBase + tid] = mp;
    bi[tid] = (float)(mp / DIM1);
    bj[tid] = (float)(mp % DIM1);
  }
  __syncthreads();

  // Gaussian-weighted loss scan over the LDS distance tile.
  // Thread (row = tid%16, seg = tid/16) covers GRID_G/16 = 144 consecutive cols.
  // colBase = seg*144 is a multiple of DIM1=48 -> (gi,gj) tracked incrementally.
  {
    const int row = tid & 15;
    const int seg = tid >> 4;
    const float bif = bi[row], bjf = bj[row];
    const float* dp = dl + row * DSTRIDE + seg * (GRID_G / 16);
    float gi = (float)(seg * 3);   // (seg*144)/48
    float gj = 0.0f;
    int jc = 0;
    float part = 0.0f;
    for (int c = 0; c < GRID_G / 16; ++c) {
      const float di = bif - gi;
      const float dj = bjf - gj;
      // exp(-gd2) = exp2(-gd2 * log2(e)) : single v_exp_f32
      part += __builtin_amdgcn_exp2f(-(di * di + dj * dj) * 1.4426950408889634f) * dp[c];
      gj += 1.0f;
      if (++jc == DIM1) { jc = 0; gj = 0.0f; gi += 1.0f; }
    }
    rs[tid] = part;
  }
  __syncthreads();
  for (int st = 128; st > 0; st >>= 1) {
    if (tid < st) rs[tid] += rs[tid + st];
    __syncthreads();
  }
  if (tid == 0) atomicAdd(lsum, rs[0]);
}

// ---- gather BMU weight vectors (full f32 precision from original w) ----
__global__ void som_gather(const float* __restrict__ w, const int* __restrict__ minpos,
                           float* __restrict__ out) {
  const int b = blockIdx.x, f = threadIdx.x;
  const int p = minpos[b];
  out[(size_t)b * FEAT + f] = w[(size_t)f * GRID_G + p];
}

__global__ void som_finalize(const float* __restrict__ lsum, float* __restrict__ out) {
  out[(size_t)BATCH * FEAT] = lsum[0] / (float)((size_t)BATCH * GRID_G);
}

extern "C" void kernel_launch(void* const* d_in, const int* in_sizes, int n_in,
                              void* d_out, int out_size, void* d_ws, size_t ws_size,
                              hipStream_t stream) {
  (void)in_sizes; (void)n_in; (void)out_size; (void)ws_size;
  const float* x = (const float*)d_in[0];   // (B, F)
  const float* w = (const float*)d_in[1];   // (F, G)
  float* out = (float*)d_out;               // closest (B*F) then Lsom (1)

  char* ws = (char*)d_ws;
  size_t off = 0;
  unsigned short* xb = (unsigned short*)(ws + off); off += (size_t)BATCH * FEAT * 2;  // 8.39 MB
  unsigned short* wt = (unsigned short*)(ws + off); off += (size_t)GRID_G * FEAT * 2; // 0.59 MB
  float* x2     = (float*)(ws + off); off += (size_t)BATCH * 4;
  float* w2     = (float*)(ws + off); off += (size_t)GRID_G * 4;
  int*   minpos = (int*)(ws + off);   off += (size_t)BATCH * 4;
  float* lsum   = (float*)(ws + off); off += 16;

  som_prep_x<<<BATCH, FEAT, 0, stream>>>(x, xb, x2);
  som_prep_w<<<GRID_G, FEAT, 0, stream>>>(w, wt, w2);
  som_zero<<<1, 1, 0, stream>>>(lsum);
  const size_t dynlds = (size_t)ROWS_PER_BLK * DSTRIDE * sizeof(float);  // ~147.7 KB
  som_fused<<<BATCH / ROWS_PER_BLK, 256, dynlds, stream>>>(xb, wt, x2, w2, minpos, lsum);
  som_gather<<<BATCH, FEAT, 0, stream>>>(w, minpos, out);
  som_finalize<<<1, 1, 0, stream>>>(lsum, out);
}
